// TPAAttention_58712202936554
// MI455X (gfx1250) — matrix operand order repaired and checked by reading, not compiled, not run on hardware
//
#include <hip/hip_runtime.h>
#include <math.h>

typedef __attribute__((ext_vector_type(8)))  _Float16 v8h;
typedef __attribute__((ext_vector_type(16))) _Float16 v16h;
typedef __attribute__((ext_vector_type(8)))  float    v8f;

#define D_MODEL   1024
#define NUM_HEADS 16
#define HEAD_DIM  64
#define SEQ_LEN   2048
#define BATCH     4
#define ROWS      (BATCH * SEQ_LEN)   // 8192

// ---------------- helpers ----------------

__device__ __forceinline__ v16h cat8(v8h lo, v8h hi) {
    v16h r;
#pragma unroll
    for (int i = 0; i < 8; ++i) { r[i] = lo[i]; r[i + 8] = hi[i]; }
    return r;
}

__device__ __forceinline__ v8f vzero8() {
    v8f z;
#pragma unroll
    for (int i = 0; i < 8; ++i) z[i] = 0.0f;
    return z;
}

__device__ __forceinline__ v8f wmma16(v16h a, v16h b, v8f c) {
    // D = A(16x32 f16) * B(32x16 f16) + C(16x16 f32)
    return __builtin_amdgcn_wmma_f32_16x16x32_f16(
        false, a, false, b, (short)0, c, false, false);
}

// Low 32 bits of a generic pointer into LDS = LDS byte offset (flat aperture
// maps ADDR[31:0] to the wave's LDS allocation).
__device__ __forceinline__ unsigned lds_off(const void* p) {
    return (unsigned)(unsigned long long)p;
}

// CDNA5 async copy: per-lane 16B global -> LDS, tracked by ASYNCcnt.
__device__ __forceinline__ void async_copy_b128(unsigned lds, const void* gptr) {
    asm volatile("global_load_async_to_lds_b128 %0, %1, off"
                 :: "v"(lds), "v"(gptr) : "memory");
}

// Stage a 64(n) x 32(k) f16 B tile into LDS (row stride 40 halves = 80B).
// 256 16B chunks, 128 threads -> 2 async ops per thread.
__device__ __forceinline__ void stage_B(const _Float16* __restrict__ Wk,
                                        unsigned ldsbase, int tid) {
    const int c0 = tid, c1 = tid + 128;            // chunk ids
    const void* g0 = Wk + (size_t)(c0 >> 2) * D_MODEL + (c0 & 3) * 8;
    const void* g1 = Wk + (size_t)(c1 >> 2) * D_MODEL + (c1 & 3) * 8;
    unsigned l0 = ldsbase + (unsigned)(((c0 >> 2) * 40 + (c0 & 3) * 8) * 2);
    unsigned l1 = ldsbase + (unsigned)(((c1 >> 2) * 40 + (c1 & 3) * 8) * 2);
    async_copy_b128(l0, g0);
    async_copy_b128(l1, g1);
}

// ---------------- kernel 1: LayerNorm (f32 in -> f16 out) ----------------

__global__ __launch_bounds__(256) void ln_kernel(
    const float* __restrict__ x, const float* __restrict__ g,
    const float* __restrict__ bta, _Float16* __restrict__ xln)
{
    const int row = blockIdx.x;
    const int tid = threadIdx.x;
    const float* xr = x + (size_t)row * D_MODEL;

    float v[4];
    float sum = 0.0f;
#pragma unroll
    for (int j = 0; j < 4; ++j) { v[j] = xr[tid + 256 * j]; sum += v[j]; }

    __shared__ float sh[256];
    sh[tid] = sum;
    __syncthreads();
    for (int s = 128; s > 0; s >>= 1) { if (tid < s) sh[tid] += sh[tid + s]; __syncthreads(); }
    const float mu = sh[0] * (1.0f / D_MODEL);
    __syncthreads();

    float sq = 0.0f;
#pragma unroll
    for (int j = 0; j < 4; ++j) { float d = v[j] - mu; sq += d * d; }
    sh[tid] = sq;
    __syncthreads();
    for (int s = 128; s > 0; s >>= 1) { if (tid < s) sh[tid] += sh[tid + s]; __syncthreads(); }
    const float rs = rsqrtf(sh[0] * (1.0f / D_MODEL) + 1e-5f);

#pragma unroll
    for (int j = 0; j < 4; ++j) {
        int c = tid + 256 * j;
        xln[(size_t)row * D_MODEL + c] = (_Float16)((v[j] - mu) * rs * g[c] + bta[c]);
    }
}

// ---------------- kernel 2: weights f32 -> f16 ----------------

__global__ __launch_bounds__(256) void cvt_kernel(
    const float* __restrict__ q, const float* __restrict__ k,
    const float* __restrict__ v, const float* __restrict__ o,
    _Float16* __restrict__ qh, _Float16* __restrict__ kh,
    _Float16* __restrict__ vh, _Float16* __restrict__ oh)
{
    const int i = blockIdx.x * 256 + threadIdx.x;
    const float* s; _Float16* d;
    switch (blockIdx.y) {
        case 0:  s = q; d = qh; break;
        case 1:  s = k; d = kh; break;
        case 2:  s = v; d = vh; break;
        default: s = o; d = oh; break;
    }
    d[i] = (_Float16)s[i];
}

// ---------------- kernel 3: QKV projection GEMM + rotary ----------------
// 128-thread block = 4 waves. Block tile: M=128 (32/wave), N=64.
// B (64x32) staged in LDS via async global->LDS copies, double-buffered.
// Q,K stored [B,H,S,d] with rotary; V stored transposed [B,H,d,S].

__global__ __launch_bounds__(128) void qkv_kernel(
    const _Float16* __restrict__ xln,
    const _Float16* __restrict__ qwh, const _Float16* __restrict__ kwh,
    const _Float16* __restrict__ vwh,
    _Float16* __restrict__ Qd, _Float16* __restrict__ Kd,
    _Float16* __restrict__ Vtd)
{
    __shared__ __align__(16) _Float16 Bbuf[2][64][40];

    const int tid  = threadIdx.x;
    const int w    = tid >> 5;
    const int lane = tid & 31;
    const int lm   = lane & 15;
    const int half = (lane >> 4) & 1;

    const int nt = blockIdx.y;
    const int z  = blockIdx.z;
    const _Float16* W = (z == 0) ? qwh : (z == 1) ? kwh : vwh;
    const _Float16* Wblk = W + (size_t)(nt * 64) * D_MODEL;

    const int mbase = blockIdx.x * 128 + w * 32;
    const _Float16* arow0 = xln + (size_t)(mbase + lm) * D_MODEL;
    const _Float16* arow1 = arow0 + (size_t)16 * D_MODEL;

    const unsigned lb0 = lds_off(&Bbuf[0][0][0]);
    const unsigned lb1 = lds_off(&Bbuf[1][0][0]);

    v8f acc0[4] = { vzero8(), vzero8(), vzero8(), vzero8() };
    v8f acc1[4] = { vzero8(), vzero8(), vzero8(), vzero8() };

    stage_B(Wblk, lb0, tid);

    for (int k0 = 0; k0 < D_MODEL; k0 += 32) {
        const int cur = (k0 >> 5) & 1;
        const bool more = (k0 + 32) < D_MODEL;
        if (more) {
            stage_B(Wblk + k0 + 32, cur ? lb0 : lb1, tid);
            asm volatile("s_wait_asynccnt 2" ::: "memory");  // cur buffer done
        } else {
            asm volatile("s_wait_asynccnt 0" ::: "memory");
        }
        __syncthreads();

        v16h A0 = cat8(*(const v8h*)(arow0 + k0 + half * 16),
                       *(const v8h*)(arow0 + k0 + half * 16 + 8));
        v16h A1 = cat8(*(const v8h*)(arow1 + k0 + half * 16),
                       *(const v8h*)(arow1 + k0 + half * 16 + 8));
#pragma unroll
        for (int t = 0; t < 4; ++t) {
            v16h B = *(const v16h*)&Bbuf[cur][t * 16 + lm][half * 16];
            acc0[t] = wmma16(A0, B, acc0[t]);
            acc1[t] = wmma16(A1, B, acc1[t]);
        }
        __syncthreads();   // reads done before next overwrite of this buffer
    }

    const int b = mbase >> 11;                     // rows per batch = 2048
    const int sbase = mbase & (SEQ_LEN - 1);

#pragma unroll
    for (int u = 0; u < 2; ++u) {
        const v8f* acc = u ? acc1 : acc0;
#pragma unroll
        for (int t = 0; t < 4; ++t) {
            int n = nt * 64 + t * 16 + lm;
            int h = n >> 6, dd = n & 63;
            if (z == 2) {
                v8h pack;
#pragma unroll
                for (int r = 0; r < 8; ++r) pack[r] = (_Float16)acc[t][r];
                *(v8h*)(Vtd + ((size_t)(b * NUM_HEADS + h) * HEAD_DIM + dd) * SEQ_LEN
                        + sbase + u * 16 + half * 8) = pack;
            } else {
                _Float16* dst = (z == 0) ? Qd : Kd;
                // inv_freq = 10000^(-(dd&31)/32)
                float invf = __expf(-(float)(dd & 31) * 0.28782313662425572f);
#pragma unroll
                for (int r = 0; r < 8; ++r) {
                    int s = sbase + u * 16 + half * 8 + r;
                    float ang = (float)s * invf;
                    float e = (dd < 32) ? sinf(ang) : cosf(ang);
                    dst[((size_t)(b * NUM_HEADS + h) * SEQ_LEN + s) * HEAD_DIM + dd] =
                        (_Float16)(acc[t][r] * e);
                }
            }
        }
    }
}

// ---------------- kernel 4: causal flash attention ----------------
// 128-thread block = 4 waves, 4 consecutive q-tiles (share K/V via WGP$).
// Per-wave LDS slice for the C-layout -> A-layout P transpose; LDS ops of a
// wave are in-order (DScnt), so no cross-wave barrier is needed (trip counts
// differ per wave under the causal loop).

__global__ __launch_bounds__(128) void attn_kernel(
    const _Float16* __restrict__ Qd, const _Float16* __restrict__ Kd,
    const _Float16* __restrict__ Vtd, const unsigned char* __restrict__ kpm,
    _Float16* __restrict__ ao)
{
    __shared__ __align__(16) _Float16 Pl[4][16][40];

    const int tid  = threadIdx.x;
    const int w    = tid >> 5;
    const int lane = tid & 31;
    const int lm   = lane & 15;
    const int half = (lane >> 4) & 1;

    const int qt = blockIdx.x * 4 + w;
    const int h  = blockIdx.y;
    const int b  = blockIdx.z;

    const _Float16* Qb = Qd  + ((size_t)(b * NUM_HEADS + h) * SEQ_LEN) * HEAD_DIM;
    const _Float16* Kb = Kd  + ((size_t)(b * NUM_HEADS + h) * SEQ_LEN) * HEAD_DIM;
    const _Float16* Vb = Vtd + ((size_t)(b * NUM_HEADS + h) * HEAD_DIM) * SEQ_LEN;

    const int qrow = qt * 16 + lm;
    v16h Aq0 = cat8(*(const v8h*)(Qb + (size_t)qrow * HEAD_DIM + half * 16),
                    *(const v8h*)(Qb + (size_t)qrow * HEAD_DIM + half * 16 + 8));
    v16h Aq1 = cat8(*(const v8h*)(Qb + (size_t)qrow * HEAD_DIM + 32 + half * 16),
                    *(const v8h*)(Qb + (size_t)qrow * HEAD_DIM + 32 + half * 16 + 8));

    v8f acc0 = vzero8(), acc1 = vzero8(), acc2 = vzero8(), acc3 = vzero8();
    float mrow[8], lrow[8];
#pragma unroll
    for (int r = 0; r < 8; ++r) { mrow[r] = -3.0e38f; lrow[r] = 0.0f; }

    const int rowbase = qt * 16 + half * 8;
    const int kend = qt * 16 + 16;

    for (int kb = 0; kb < kend; kb += 32) {
        // scores: 16 q x 32 keys over d=64 (2 WMMA k-steps per key half)
        v8f sc0 = vzero8(), sc1 = vzero8();
#pragma unroll
        for (int dstep = 0; dstep < 2; ++dstep) {
            v16h a = (dstep == 0) ? Aq0 : Aq1;
            const _Float16* k0p = Kb + (size_t)(kb + lm)      * HEAD_DIM + dstep * 32 + half * 16;
            const _Float16* k1p = Kb + (size_t)(kb + 16 + lm) * HEAD_DIM + dstep * 32 + half * 16;
            sc0 = wmma16(a, *(const v16h*)k0p, sc0);
            sc1 = wmma16(a, *(const v16h*)k1p, sc1);
        }

        const int c0 = kb + lm, c1 = kb + 16 + lm;
        const float pm0 = kpm[b * SEQ_LEN + c0] ? -3.0e38f : 0.0f;
        const float pm1 = kpm[b * SEQ_LEN + c1] ? -3.0e38f : 0.0f;

        float p0[8], p1[8];
#pragma unroll
        for (int r = 0; r < 8; ++r) {
            int rq = rowbase + r;
            float v0 = sc0[r] * 0.125f + pm0; if (c0 > rq) v0 = -3.0e38f;
            float v1 = sc1[r] * 0.125f + pm1; if (c1 > rq) v1 = -3.0e38f;
            float cm = fmaxf(v0, v1);
#pragma unroll
            for (int m = 1; m < 16; m <<= 1) cm = fmaxf(cm, __shfl_xor(cm, m, 32));
            float mn  = fmaxf(mrow[r], cm);
            float scl = __expf(mrow[r] - mn);
            float e0 = __expf(v0 - mn), e1 = __expf(v1 - mn);
            float ps = e0 + e1;
#pragma unroll
            for (int m = 1; m < 16; m <<= 1) ps += __shfl_xor(ps, m, 32);
            lrow[r] = lrow[r] * scl + ps;
            mrow[r] = mn;
            acc0[r] *= scl; acc1[r] *= scl; acc2[r] *= scl; acc3[r] *= scl;
            p0[r] = e0; p1[r] = e1;
        }

        // C-layout P -> LDS -> A-layout (same wave; LDS is in-order per wave)
#pragma unroll
        for (int r = 0; r < 8; ++r) {
            Pl[w][r + half * 8][lm]      = (_Float16)p0[r];
            Pl[w][r + half * 8][16 + lm] = (_Float16)p1[r];
        }
        asm volatile("" ::: "memory");
        v16h Ap = cat8(*(const v8h*)&Pl[w][lm][half * 16],
                       *(const v8h*)&Pl[w][lm][half * 16 + 8]);
        asm volatile("" ::: "memory");

        // acc += P(16x32) @ V(32x64); B columns are Vt rows (contiguous in k)
        acc0 = wmma16(Ap, *(const v16h*)(Vb + (size_t)(lm)      * SEQ_LEN + kb + half * 16), acc0);
        acc1 = wmma16(Ap, *(const v16h*)(Vb + (size_t)(16 + lm) * SEQ_LEN + kb + half * 16), acc1);
        acc2 = wmma16(Ap, *(const v16h*)(Vb + (size_t)(32 + lm) * SEQ_LEN + kb + half * 16), acc2);
        acc3 = wmma16(Ap, *(const v16h*)(Vb + (size_t)(48 + lm) * SEQ_LEN + kb + half * 16), acc3);
    }

#pragma unroll
    for (int r = 0; r < 8; ++r) {
        float inv = 1.0f / lrow[r];
        int s = rowbase + r;
        size_t base = ((size_t)(b * SEQ_LEN + s)) * D_MODEL + h * HEAD_DIM;
        ao[base + 0  + lm] = (_Float16)(acc0[r] * inv);
        ao[base + 16 + lm] = (_Float16)(acc1[r] * inv);
        ao[base + 32 + lm] = (_Float16)(acc2[r] * inv);
        ao[base + 48 + lm] = (_Float16)(acc3[r] * inv);
    }
}

// ---------------- kernel 5: output projection + bias + residual ----------------
// Same blocking as qkv: 4 waves, M=128/block, N=64, async-staged B.

__global__ __launch_bounds__(128) void proj_kernel(
    const _Float16* __restrict__ ao, const _Float16* __restrict__ owh,
    const float* __restrict__ x, const float* __restrict__ ob,
    float* __restrict__ out)
{
    __shared__ __align__(16) _Float16 Bbuf[2][64][40];

    const int tid  = threadIdx.x;
    const int w    = tid >> 5;
    const int lane = tid & 31;
    const int lm   = lane & 15;
    const int half = (lane >> 4) & 1;

    const int nt = blockIdx.y;
    const _Float16* Wblk = owh + (size_t)(nt * 64) * D_MODEL;

    const int mbase = blockIdx.x * 128 + w * 32;
    const _Float16* arow0 = ao + (size_t)(mbase + lm) * D_MODEL;
    const _Float16* arow1 = arow0 + (size_t)16 * D_MODEL;

    const unsigned lb0 = lds_off(&Bbuf[0][0][0]);
    const unsigned lb1 = lds_off(&Bbuf[1][0][0]);

    v8f acc0[4] = { vzero8(), vzero8(), vzero8(), vzero8() };
    v8f acc1[4] = { vzero8(), vzero8(), vzero8(), vzero8() };

    stage_B(Wblk, lb0, tid);

    for (int k0 = 0; k0 < D_MODEL; k0 += 32) {
        const int cur = (k0 >> 5) & 1;
        const bool more = (k0 + 32) < D_MODEL;
        if (more) {
            stage_B(Wblk + k0 + 32, cur ? lb0 : lb1, tid);
            asm volatile("s_wait_asynccnt 2" ::: "memory");
        } else {
            asm volatile("s_wait_asynccnt 0" ::: "memory");
        }
        __syncthreads();

        v16h A0 = cat8(*(const v8h*)(arow0 + k0 + half * 16),
                       *(const v8h*)(arow0 + k0 + half * 16 + 8));
        v16h A1 = cat8(*(const v8h*)(arow1 + k0 + half * 16),
                       *(const v8h*)(arow1 + k0 + half * 16 + 8));
#pragma unroll
        for (int t = 0; t < 4; ++t) {
            v16h B = *(const v16h*)&Bbuf[cur][t * 16 + lm][half * 16];
            acc0[t] = wmma16(A0, B, acc0[t]);
            acc1[t] = wmma16(A1, B, acc1[t]);
        }
        __syncthreads();
    }

#pragma unroll
    for (int u = 0; u < 2; ++u) {
        const v8f* acc = u ? acc1 : acc0;
#pragma unroll
        for (int t = 0; t < 4; ++t) {
            int n = nt * 64 + t * 16 + lm;
            float bn = ob[n];
#pragma unroll
            for (int r = 0; r < 8; ++r) {
                size_t M = (size_t)mbase + u * 16 + half * 8 + r;
                out[M * D_MODEL + n] = x[M * D_MODEL + n] + bn + acc[t][r];
            }
        }
    }
}

// ---------------- host launch ----------------

extern "C" void kernel_launch(void* const* d_in, const int* in_sizes, int n_in,
                              void* d_out, int out_size, void* d_ws, size_t ws_size,
                              hipStream_t stream)
{
    (void)in_sizes; (void)n_in; (void)out_size; (void)ws_size;

    const float* x   = (const float*)d_in[0];
    const unsigned char* kpm = (const unsigned char*)d_in[1];
    const float* qw  = (const float*)d_in[2];
    const float* kw  = (const float*)d_in[3];
    const float* vw  = (const float*)d_in[4];
    const float* ow  = (const float*)d_in[5];
    const float* ob  = (const float*)d_in[6];
    const float* lng = (const float*)d_in[7];
    const float* lnb = (const float*)d_in[8];
    float* out = (float*)d_out;

    char* ws = (char*)d_ws;
    const size_t SZ_ROWMAT = (size_t)ROWS * D_MODEL * sizeof(_Float16);     // 16 MiB
    const size_t SZ_W      = (size_t)D_MODEL * D_MODEL * sizeof(_Float16);  // 2 MiB

    _Float16* xln = (_Float16*)(ws);
    _Float16* qwh = (_Float16*)(ws + SZ_ROWMAT);
    _Float16* kwh = (_Float16*)(ws + SZ_ROWMAT + 1 * SZ_W);
    _Float16* vwh = (_Float16*)(ws + SZ_ROWMAT + 2 * SZ_W);
    _Float16* owh = (_Float16*)(ws + SZ_ROWMAT + 3 * SZ_W);
    _Float16* Qd  = (_Float16*)(ws + SZ_ROWMAT + 4 * SZ_W);
    _Float16* Kd  = (_Float16*)(ws + 2 * SZ_ROWMAT + 4 * SZ_W);
    _Float16* Vtd = (_Float16*)(ws + 3 * SZ_ROWMAT + 4 * SZ_W);
    _Float16* ao  = (_Float16*)(ws + 4 * SZ_ROWMAT + 4 * SZ_W);

    ln_kernel<<<ROWS, 256, 0, stream>>>(x, lng, lnb, xln);
    cvt_kernel<<<dim3(D_MODEL * D_MODEL / 256, 4), 256, 0, stream>>>(
        qw, kw, vw, ow, qwh, kwh, vwh, owh);
    qkv_kernel<<<dim3(ROWS / 128, D_MODEL / 64, 3), 128, 0, stream>>>(
        xln, qwh, kwh, vwh, Qd, Kd, Vtd);
    attn_kernel<<<dim3(SEQ_LEN / 64, NUM_HEADS, BATCH), 128, 0, stream>>>(
        Qd, Kd, Vtd, kpm, ao);
    proj_kernel<<<dim3(ROWS / 128, D_MODEL / 64), 128, 0, stream>>>(
        ao, owh, x, ob, out);
}
